// ContrastiveLoss_89507118449313
// MI455X (gfx1250) — matrix operand order repaired and checked by reading, not compiled, not run on hardware
//
#include <hip/hip_runtime.h>
#include <hip/hip_bf16.h>

typedef _Float16 v4h  __attribute__((ext_vector_type(4)));
typedef _Float16 v8h  __attribute__((ext_vector_type(8)));
typedef _Float16 v16h __attribute__((ext_vector_type(16)));
typedef float    v8f  __attribute__((ext_vector_type(8)));
typedef int      v4i  __attribute__((ext_vector_type(4)));

#define TILE      128
#define KC        32
#define LDS_PITCH 40        // halves per LDS row: 32 + 8 pad (80B; rows stay 16B-aligned)

// exp(sim/T) == exp2(sim * log2(e)/T)
constexpr float kExp2Scale = 1.4426950408889634f / 0.07f;

#if defined(__HIP_DEVICE_COMPILE__) && defined(__gfx1250__) && \
    __has_builtin(__builtin_amdgcn_global_load_async_to_lds_b128)
#define HAVE_ASYNC_LDS 1
#else
#define HAVE_ASYNC_LDS 0
#endif

#if HAVE_ASYNC_LDS
// 16-byte async DMA: global -> LDS, bypassing VGPRs, tracked by ASYNCcnt.
__device__ __forceinline__ void async_copy_b128(const _Float16* gsrc, _Float16* ldst) {
  __builtin_amdgcn_global_load_async_to_lds_b128(
      (__attribute__((address_space(1))) v4i*)gsrc,
      (__attribute__((address_space(3))) v4i*)ldst,
      0, 0);
}
#endif

// ---------------- Kernel 0: zero the per-row accumulators ----------------
__global__ void zero_accum_kernel(float* __restrict__ p, int n) {
  int i = blockIdx.x * blockDim.x + threadIdx.x;
  if (i < n) p[i] = 0.0f;
}

// ---------------- Kernel 1: L2-normalize rows, emit f16 ----------------
// grid = N blocks, 128 threads; D = 512 (4 floats / thread)
__global__ __launch_bounds__(128)
void normalize_f16_kernel(const float* __restrict__ H, _Float16* __restrict__ Hh, int D) {
  const int row = blockIdx.x;
  const int tid = threadIdx.x;
  const float4 x = ((const float4*)(H + (size_t)row * D))[tid];
  float ss = x.x * x.x + x.y * x.y + x.z * x.z + x.w * x.w;
  #pragma unroll
  for (int m = 1; m < 32; m <<= 1) ss += __shfl_xor(ss, m, 32);

  __shared__ float warp_ss[4];
  __shared__ float total_ss;
  if ((tid & 31) == 0) warp_ss[tid >> 5] = ss;
  __syncthreads();
  if (tid == 0) total_ss = warp_ss[0] + warp_ss[1] + warp_ss[2] + warp_ss[3];
  __syncthreads();

  const float inv = 1.0f / fmaxf(sqrtf(total_ss), 1e-12f);
  v4h o;
  o[0] = (_Float16)(x.x * inv);
  o[1] = (_Float16)(x.y * inv);
  o[2] = (_Float16)(x.z * inv);
  o[3] = (_Float16)(x.w * inv);
  *(v4h*)(Hh + (size_t)row * D + tid * 4) = o;
}

// ---------------- Kernel 2: fused WMMA GEMM tile + exp + masked row sums ----------------
// grid = (N/128, N/128), 256 threads = 8 waves (4 M x 2 N), wave covers 32x64.
// K is double-buffered through LDS in chunks of 32, staged with async-to-LDS copies.
__global__ __launch_bounds__(256)
void simcon_tile_kernel(const _Float16* __restrict__ Hh,
                        const int* __restrict__ labels,
                        float* __restrict__ rowTot,
                        float* __restrict__ rowPos,
                        float* __restrict__ rowCnt,
                        int D) {
  __shared__ __align__(16) _Float16 As[2][TILE * LDS_PITCH];
  __shared__ __align__(16) _Float16 Bs[2][TILE * LDS_PITCH];
  __shared__ int   labR[TILE], labC[TILE];
  __shared__ float shTot[TILE], shPos[TILE], shCnt[TILE];

  const int tid   = threadIdx.x;
  const int tileM = blockIdx.y, tileN = blockIdx.x;
  const int rowBaseA = tileM * TILE, rowBaseB = tileN * TILE;
  const int wave = tid >> 5, lane = tid & 31;
  const int l = lane & 15, hi = lane >> 4;
  const int waveM = wave >> 1, waveN = wave & 1;

  if (tid < TILE) {
    labR[tid]  = labels[rowBaseA + tid];
    labC[tid]  = labels[rowBaseB + tid];
    shTot[tid] = 0.0f; shPos[tid] = 0.0f; shCnt[tid] = 0.0f;
  }

  // ---- K-chunk staging: 128 rows x 32 halves per matrix = 512 x 16B segments ----
  auto stage_chunk = [&](int kc, int buf) {
    #pragma unroll
    for (int it = 0; it < 2; ++it) {
      const int e  = tid + it * 256;   // 0..511
      const int r  = e >> 2;           // 0..127
      const int c8 = (e & 3) * 8;      // 0,8,16,24
#if HAVE_ASYNC_LDS
      async_copy_b128(&Hh[(size_t)(rowBaseA + r) * D + kc + c8],
                      &As[buf][r * LDS_PITCH + c8]);
      async_copy_b128(&Hh[(size_t)(rowBaseB + r) * D + kc + c8],
                      &Bs[buf][r * LDS_PITCH + c8]);
#else
      *(v8h*)&As[buf][r * LDS_PITCH + c8] =
          *(const v8h*)&Hh[(size_t)(rowBaseA + r) * D + kc + c8];
      *(v8h*)&Bs[buf][r * LDS_PITCH + c8] =
          *(const v8h*)&Hh[(size_t)(rowBaseB + r) * D + kc + c8];
#endif
    }
  };

  v8f acc[2][4];
  #pragma unroll
  for (int tm = 0; tm < 2; ++tm)
    #pragma unroll
    for (int tn = 0; tn < 4; ++tn)
      #pragma unroll
      for (int r = 0; r < 8; ++r) acc[tm][tn][r] = 0.0f;

  const int nsteps = D / KC;
  stage_chunk(0, 0);

  for (int step = 0; step < nsteps; ++step) {
    const int cur = step & 1;
#if HAVE_ASYNC_LDS
    asm volatile("s_wait_asynccnt 0" ::: "memory");
#endif
    __syncthreads();   // buf[cur] fully written; buf[cur^1] fully consumed
    if (step + 1 < nsteps) stage_chunk((step + 1) * KC, cur ^ 1);

    v16h afr[2], bfr[4];
    // A fragment (16x32): lanes 0-15 -> K = 0..7, 16..23 ; lanes 16-31 -> +8
    #pragma unroll
    for (int tm = 0; tm < 2; ++tm) {
      const int rrow = waveM * 32 + tm * 16 + l;
      const int kb   = hi * 8;
      v8h lo8 = *(const v8h*)&As[cur][rrow * LDS_PITCH + kb];
      v8h hi8 = *(const v8h*)&As[cur][rrow * LDS_PITCH + kb + 16];
      #pragma unroll
      for (int i = 0; i < 8; ++i) { afr[tm][i] = lo8[i]; afr[tm][i + 8] = hi8[i]; }
    }
    // B fragment (32x16): lanes 0-15 -> K = 0..7, 16..23 ; lanes 16-31 -> +8
    #pragma unroll
    for (int tn = 0; tn < 4; ++tn) {
      const int brow = waveN * 64 + tn * 16 + l;
      const int kb   = hi * 8;
      v8h lo8 = *(const v8h*)&Bs[cur][brow * LDS_PITCH + kb];
      v8h hi8 = *(const v8h*)&Bs[cur][brow * LDS_PITCH + kb + 16];
      #pragma unroll
      for (int i = 0; i < 8; ++i) { bfr[tn][i] = lo8[i]; bfr[tn][i + 8] = hi8[i]; }
    }
    #pragma unroll
    for (int tm = 0; tm < 2; ++tm)
      #pragma unroll
      for (int tn = 0; tn < 4; ++tn)
        acc[tm][tn] = __builtin_amdgcn_wmma_f32_16x16x32_f16(
            false, afr[tm], false, bfr[tn], (short)0, acc[tm][tn], false, false);
  }
  __syncthreads();

  // Epilogue: exp + label mask + per-row reduction.
  // C/D layout: element r -> M = r + hi*8, N = l (16 lanes share a row).
  #pragma unroll
  for (int tm = 0; tm < 2; ++tm) {
    #pragma unroll
    for (int r = 0; r < 8; ++r) {
      const int rowLocal = waveM * 32 + tm * 16 + hi * 8 + r;
      const int gi = rowBaseA + rowLocal;
      const int li = labR[rowLocal];
      float t = 0.0f, p = 0.0f, c = 0.0f;
      #pragma unroll
      for (int tn = 0; tn < 4; ++tn) {
        const int colLocal = waveN * 64 + tn * 16 + l;
        const int gj = rowBaseB + colLocal;
        const float e = exp2f(acc[tm][tn][r] * kExp2Scale);
        t += e;
        if (labC[colLocal] == li && gi != gj) { p += e; c += 1.0f; }
      }
      #pragma unroll
      for (int m = 1; m < 16; m <<= 1) {
        t += __shfl_xor(t, m, 32);
        p += __shfl_xor(p, m, 32);
        c += __shfl_xor(c, m, 32);
      }
      if (l == 0) {
        atomicAdd(&shTot[rowLocal], t);
        atomicAdd(&shPos[rowLocal], p);
        atomicAdd(&shCnt[rowLocal], c);
      }
    }
  }
  __syncthreads();
  if (tid < TILE) {
    atomicAdd(&rowTot[rowBaseA + tid], shTot[tid]);
    atomicAdd(&rowPos[rowBaseA + tid], shPos[tid]);
    atomicAdd(&rowCnt[rowBaseA + tid], shCnt[tid]);
  }
}

// ---------------- Kernel 3: final scalar loss ----------------
__global__ __launch_bounds__(256)
void final_loss_kernel(const float* __restrict__ rowTot,
                       const float* __restrict__ rowPos,
                       const float* __restrict__ rowCnt,
                       float* __restrict__ out, int N) {
  __shared__ float red[256];
  float s = 0.0f;
  for (int i = threadIdx.x; i < N; i += 256) {
    const float pos_sum = rowPos[i] / (rowCnt[i] + 1e-9f);
    s += logf(pos_sum / rowTot[i]);
  }
  red[threadIdx.x] = s;
  __syncthreads();
  #pragma unroll
  for (int stride = 128; stride > 0; stride >>= 1) {
    if (threadIdx.x < stride) red[threadIdx.x] += red[threadIdx.x + stride];
    __syncthreads();
  }
  if (threadIdx.x == 0) out[0] = -red[0] / (float)N;
}

extern "C" void kernel_launch(void* const* d_in, const int* in_sizes, int n_in,
                              void* d_out, int out_size, void* d_ws, size_t ws_size,
                              hipStream_t stream) {
  const float* hidden = (const float*)d_in[0];
  const int*   labels = (const int*)d_in[1];   // harness passes integers as int32
  float* out = (float*)d_out;

  const int N = in_sizes[1];                   // 8192
  const int D = in_sizes[0] / N;               // 512

  // workspace layout
  _Float16* Hh = (_Float16*)d_ws;                              // N*D halves
  float* rowTot = (float*)((char*)d_ws + (size_t)N * D * sizeof(_Float16));
  float* rowPos = rowTot + N;
  float* rowCnt = rowPos + N;

  // zero accumulators (harness does not re-poison between replays)
  {
    const int n = 3 * N;
    zero_accum_kernel<<<(n + 255) / 256, 256, 0, stream>>>(rowTot, n);
  }

  // normalize + f16 convert
  normalize_f16_kernel<<<N, 128, 0, stream>>>(hidden, Hh, D);

  // fused WMMA similarity + exp + masked row sums
  dim3 grid(N / TILE, N / TILE);
  simcon_tile_kernel<<<grid, 256, 0, stream>>>(Hh, labels, rowTot, rowPos, rowCnt, D);

  // scalar loss
  final_loss_kernel<<<1, 256, 0, stream>>>(rowTot, rowPos, rowCnt, out, N);
}
